// FlexBertPaddedRopeAttention_43731357008514
// MI455X (gfx1250) — compile-verified, matrix operand order checked by reading, not run on hardware
//
#include <hip/hip_runtime.h>
#include <hip/hip_bf16.h>
#include <stdint.h>

// Problem constants (from reference): B=4, S=2048, D=1024, H=16, HD=64, ROT_DIM=64
#define SEQ     2048
#define DMODEL  1024
#define NHEAD   16
#define HDIM    64
#define BATCH   4
#define MTOT    (BATCH * SEQ)        // 8192 rows of the token matrix

typedef __attribute__((ext_vector_type(16))) _Float16 v16h;
typedef __attribute__((ext_vector_type(8)))  _Float16 v8h;
typedef __attribute__((ext_vector_type(4)))  _Float16 v4h;
typedef __attribute__((ext_vector_type(8)))  float    v8f;

// ---------------------------------------------------------------------------
// WMMA helpers (CDNA5 wave32, 16x16x32 f16 -> f32 accumulate)
// ---------------------------------------------------------------------------
__device__ __forceinline__ v8f wmma_f16(v16h a, v16h b, v8f c) {
  return __builtin_amdgcn_wmma_f32_16x16x32_f16(
      /*neg_a=*/false, a, /*neg_b=*/false, b,
      /*c_mod=*/(short)0, c, /*reuse_a=*/false, /*reuse_b=*/false);
}

// A/B fragment (16x32 f16) per-lane layout (ISA 7.12.2):
// lane L: row/col = L&15 ; k = (j<8 ? j : j+8) + (L>=16 ? 8 : 0)
__device__ __forceinline__ v16h frag_f16(const _Float16* p) {
  v8h lo = *(const v8h*)(p);
  v8h hi = *(const v8h*)(p + 16);
  v16h r;
#pragma unroll
  for (int j = 0; j < 8; ++j) { r[j] = lo[j]; r[j + 8] = hi[j]; }
  return r;
}

// ---------------------------------------------------------------------------
// CDNA5 async global->LDS copy (16B per lane), tracked by ASYNCcnt.
// Flat LDS addresses carry the wave-relative LDS offset in addr[31:0].
// ---------------------------------------------------------------------------
__device__ __forceinline__ void async_copy_b128(_Float16* lds_dst, const _Float16* gsrc) {
  uint32_t loff = (uint32_t)(uintptr_t)(void*)lds_dst;
  asm volatile("global_load_async_to_lds_b128 %0, %1, off"
               :: "v"(loff), "v"(gsrc)
               : "memory");
}
__device__ __forceinline__ void wait_async0() {
  asm volatile("s_wait_asynccnt 0x0" ::: "memory");
}

// ---------------------------------------------------------------------------
// Kernel P: f32 -> f16 pre-conversion (X, Wqkv, Wo), 4 elems / thread
// ---------------------------------------------------------------------------
__global__ void cvt_f16_kernel(const float* __restrict__ s,
                               _Float16* __restrict__ d, int n4) {
  int i = blockIdx.x * blockDim.x + threadIdx.x;
  if (i >= n4) return;
  const float* p = s + (size_t)i * 4;
  v4h o;
#pragma unroll
  for (int j = 0; j < 4; ++j) o[j] = (_Float16)p[j];
  *(v4h*)(d + (size_t)i * 4) = o;
}

// ---------------------------------------------------------------------------
// Kernel 0: RoPE cos/sin tables  [SEQ][32] each
// ---------------------------------------------------------------------------
__global__ void rope_table_kernel(float* __restrict__ cosT, float* __restrict__ sinT) {
  int i = blockIdx.x * blockDim.x + threadIdx.x;
  if (i >= SEQ * 32) return;
  int s = i >> 5;
  int f = i & 31;
  float inv_freq = __powf(10000.0f, -(float)(2 * f) / 64.0f);
  float ang = (float)s * inv_freq;
  cosT[i] = __cosf(ang);
  sinT[i] = __sinf(ang);
}

// ---------------------------------------------------------------------------
// Kernel 1: QKV GEMM (M=8192, N=3072, K=1024) + bias + RoPE, all-f16 operands.
// Weight tile (64 cols x 64 k) double-buffered in LDS via async global->LDS
// copies; one barrier per 64-k step; copy for step i+1 overlaps WMMAs of i.
// ---------------------------------------------------------------------------
__global__ __launch_bounds__(256) void qkv_rope_kernel(
    const _Float16* __restrict__ X, const _Float16* __restrict__ W,
    const float* __restrict__ bias,
    const float* __restrict__ cosT, const float* __restrict__ sinT,
    _Float16* __restrict__ Qb, _Float16* __restrict__ Kb,
    _Float16* __restrict__ Vt) {
  __shared__ _Float16 BT[2][64 * 64];         // 2 x 8 KB weight tiles

  const int lane = threadIdx.x & 31;
  const int wave = threadIdx.x >> 5;
  const int ln   = lane & 15;
  const int half = lane >> 4;
  const int kg   = half * 8;

  const int mtile  = blockIdx.x * 128 + wave * 16;
  const int n_base = blockIdx.y * 64;

  const _Float16* arow = X + (size_t)(mtile + ln) * DMODEL + kg;
  const int cr   = threadIdx.x >> 2;          // 0..63 : weight row
  const int cp   = (threadIdx.x & 3) * 16;    // 0,16,32,48 : k sub-offset
  const _Float16* wsrc = W + (size_t)(n_base + cr) * DMODEL + cp;

  v8f acc[4] = {v8f{}, v8f{}, v8f{}, v8f{}};

  // preload first tile
  async_copy_b128(&BT[0][cr * 64 + cp],     wsrc);
  async_copy_b128(&BT[0][cr * 64 + cp + 8], wsrc + 8);

  for (int k = 0; k < DMODEL; k += 64) {
    const int cur = (k >> 6) & 1;
    wait_async0();
    __syncthreads();
    if (k + 64 < DMODEL) {
      async_copy_b128(&BT[cur ^ 1][cr * 64 + cp],     wsrc + k + 64);
      async_copy_b128(&BT[cur ^ 1][cr * 64 + cp + 8], wsrc + k + 64 + 8);
    }
#pragma unroll
    for (int ks = 0; ks < 64; ks += 32) {
      v16h a = frag_f16(arow + k + ks);
#pragma unroll
      for (int j = 0; j < 4; ++j)
        acc[j] = wmma_f16(a, frag_f16(&BT[cur][(16 * j + ln) * 64 + ks + kg]), acc[j]);
    }
  }

  // bias (per column, same for all 8 rows of each accumulator)
#pragma unroll
  for (int j = 0; j < 4; ++j) {
    float bj = bias[n_base + 16 * j + ln];
#pragma unroll
    for (int r = 0; r < 8; ++r) acc[j][r] += bj;
  }

  const int which = n_base / DMODEL;            // 0=q 1=k 2=v
  const int h     = (n_base % DMODEL) / HDIM;   // head
  const int mrow  = mtile + half * 8;           // first row held by this lane
  const int b     = mrow >> 11;                 // batch (rows never straddle)
  const int spos  = mrow & (SEQ - 1);
  const int bh    = b * NHEAD + h;

  if (which < 2) {
    // RoPE: pairs (d, d+32) => (acc0,acc2) with f=ln, (acc1,acc3) with f=16+ln
#pragma unroll
    for (int r = 0; r < 8; ++r) {
      int s = spos + r;
      float c0 = cosT[s * 32 + ln],      s0 = sinT[s * 32 + ln];
      float c1 = cosT[s * 32 + 16 + ln], s1 = sinT[s * 32 + 16 + ln];
      float x0 = acc[0][r], x2 = acc[2][r];
      acc[0][r] = x0 * c0 - x2 * s0;
      acc[2][r] = x0 * s0 + x2 * c0;
      float x1 = acc[1][r], x3 = acc[3][r];
      acc[1][r] = x1 * c1 - x3 * s1;
      acc[3][r] = x1 * s1 + x3 * c1;
    }
    _Float16* dst = (which == 0) ? Qb : Kb;
#pragma unroll
    for (int j = 0; j < 4; ++j)
#pragma unroll
      for (int r = 0; r < 8; ++r)
        dst[((size_t)bh * SEQ + spos + r) * HDIM + 16 * j + ln] =
            (_Float16)acc[j][r];
  } else {
    // V transposed: Vt[bh][d][s]; lane's 8 rows are contiguous s positions
#pragma unroll
    for (int j = 0; j < 4; ++j) {
      int d = 16 * j + ln;
      v8h pk;
#pragma unroll
      for (int r = 0; r < 8; ++r) pk[r] = (_Float16)acc[j][r];
      *(v8h*)(Vt + ((size_t)bh * HDIM + d) * SEQ + spos) = pk;
    }
  }
}

// ---------------------------------------------------------------------------
// Kernel 2: flash attention per (b,h).  grid = (SEQ/128, B*H), 8 waves/block.
// K and V^T tiles (64-key blocks) double-buffered in LDS via async copies;
// one barrier per key block.
// ---------------------------------------------------------------------------
__global__ __launch_bounds__(256) void attn_kernel(
    const _Float16* __restrict__ Q, const _Float16* __restrict__ K,
    const _Float16* __restrict__ Vt, _Float16* __restrict__ A) {
  __shared__ _Float16 Kt[2][64 * 64];      // 2 x 8 KB  [key][d]
  __shared__ _Float16 Vs[2][64 * 64];      // 2 x 8 KB  [d][key]
  __shared__ _Float16 Pst[8][16 * 64];     // 16 KB wave-private P staging

  const int lane = threadIdx.x & 31;
  const int wave = threadIdx.x >> 5;
  const int ln   = lane & 15;
  const int half = lane >> 4;
  const int kg   = half * 8;

  const int bh = blockIdx.y;
  const int b  = bh >> 4;
  const int h  = bh & 15;
  const int q0 = blockIdx.x * 128 + wave * 16;

  const _Float16* Qp = Q  + (size_t)bh * SEQ * HDIM;
  const _Float16* Kp = K  + (size_t)bh * SEQ * HDIM;
  const _Float16* Vp = Vt + (size_t)bh * HDIM * SEQ;

  const int cr = threadIdx.x >> 2;         // 0..63
  const int cp = (threadIdx.x & 3) * 16;   // 0,16,32,48

  // Q fragments for this wave's 16 rows (d = 0..63 -> two k-steps)
  const _Float16* qrow = Qp + (size_t)(q0 + ln) * HDIM + kg;
  const v16h qa0 = frag_f16(qrow);
  const v16h qa1 = frag_f16(qrow + 32);

  v8f o[4] = {v8f{}, v8f{}, v8f{}, v8f{}};
  float rmax[8], rsum[8];
#pragma unroll
  for (int r = 0; r < 8; ++r) { rmax[r] = -1e30f; rsum[r] = 0.0f; }

  _Float16* pw = Pst[wave];
  const float scale = 0.125f;              // 1/sqrt(64)

  // stage key block kb into buffer buf (32B K + 32B V per thread)
  auto stage = [&](int buf, int kb) {
    const _Float16* ksrc = Kp + (size_t)(kb + cr) * HDIM + cp;
    async_copy_b128(&Kt[buf][cr * 64 + cp],     ksrc);
    async_copy_b128(&Kt[buf][cr * 64 + cp + 8], ksrc + 8);
    const _Float16* vsrc = Vp + (size_t)cr * SEQ + kb + cp;
    async_copy_b128(&Vs[buf][cr * 64 + cp],     vsrc);
    async_copy_b128(&Vs[buf][cr * 64 + cp + 8], vsrc + 8);
  };
  stage(0, 0);

  for (int kb = 0; kb < SEQ; kb += 64) {
    const int cur = (kb >> 6) & 1;
    wait_async0();
    __syncthreads();
    if (kb + 64 < SEQ) stage(cur ^ 1, kb + 64);

    // ---- scores: 4 tiles of 16 q x 16 keys (B fragments from LDS)
    v8f st[4];
#pragma unroll
    for (int t = 0; t < 4; ++t) {
      const _Float16* krow = &Kt[cur][(t * 16 + ln) * 64 + kg];
      v8f s = {};
      s = wmma_f16(qa0, frag_f16(krow), s);
      s = wmma_f16(qa1, frag_f16(krow + 32), s);
#pragma unroll
      for (int r = 0; r < 8; ++r) s[r] *= scale;
      st[t] = s;
    }

    // ---- row-wise max over the 64-key block (reduce across 16-lane halves)
    float bm[8], ps[8];
#pragma unroll
    for (int r = 0; r < 8; ++r)
      bm[r] = fmaxf(fmaxf(st[0][r], st[1][r]), fmaxf(st[2][r], st[3][r]));
#pragma unroll
    for (int mask = 1; mask <= 8; mask <<= 1)
#pragma unroll
      for (int r = 0; r < 8; ++r)
        bm[r] = fmaxf(bm[r], __shfl_xor(bm[r], mask, 32));

    // ---- online softmax update
#pragma unroll
    for (int r = 0; r < 8; ++r) {
      float mnew  = fmaxf(rmax[r], bm[r]);
      float alpha = __expf(rmax[r] - mnew);
      rmax[r] = mnew;
      rsum[r] *= alpha;
#pragma unroll
      for (int j = 0; j < 4; ++j) o[j][r] *= alpha;
      float psum = 0.0f;
#pragma unroll
      for (int t = 0; t < 4; ++t) {
        float p = __expf(st[t][r] - mnew);
        st[t][r] = p;
        psum += p;
      }
      ps[r] = psum;
    }
#pragma unroll
    for (int mask = 1; mask <= 8; mask <<= 1)
#pragma unroll
      for (int r = 0; r < 8; ++r) ps[r] += __shfl_xor(ps[r], mask, 32);
#pragma unroll
    for (int r = 0; r < 8; ++r) rsum[r] += ps[r];

    // ---- relayout P: C-fragment -> row-major [16 q][64 keys] (wave-private)
#pragma unroll
    for (int t = 0; t < 4; ++t)
#pragma unroll
      for (int r = 0; r < 8; ++r)
        pw[(half * 8 + r) * 64 + t * 16 + ln] = (_Float16)st[t][r];
    // wave-private region; LDS is in-order per wave

    const v16h pa0 = frag_f16(pw + ln * 64 + kg);
    const v16h pa1 = frag_f16(pw + ln * 64 + 32 + kg);

    // ---- O += P @ V  (V^T tile in LDS gives contiguous B fragments)
#pragma unroll
    for (int j = 0; j < 4; ++j) {
      const _Float16* vp = &Vs[cur][(16 * j + ln) * 64 + kg];
      o[j] = wmma_f16(pa0, frag_f16(vp), o[j]);
      o[j] = wmma_f16(pa1, frag_f16(vp + 32), o[j]);
    }
    // WMMAs consumed all ds loads of buffer `cur` (dscnt-waited) before the
    // next iteration's barrier, so the async writes to `cur^1` cannot race.
  }

  // ---- normalize and store attn output f16 into [MTOT][DMODEL]
  float inv[8];
#pragma unroll
  for (int r = 0; r < 8; ++r) inv[r] = 1.0f / rsum[r];
  const size_t rowbase = (size_t)b * SEQ + q0 + half * 8;
#pragma unroll
  for (int j = 0; j < 4; ++j)
#pragma unroll
    for (int r = 0; r < 8; ++r)
      A[(rowbase + r) * DMODEL + h * HDIM + 16 * j + ln] =
          (_Float16)(o[j][r] * inv[r]);
}

// ---------------------------------------------------------------------------
// Kernel 3: output projection  out = attn @ Wo^T + b   (M=8192,N=1024,K=1024)
// Same double-buffered async-staged structure as the QKV GEMM.
// ---------------------------------------------------------------------------
__global__ __launch_bounds__(256) void out_proj_kernel(
    const _Float16* __restrict__ Attn, const _Float16* __restrict__ W,
    const float* __restrict__ bias, float* __restrict__ out) {
  __shared__ _Float16 BT[2][64 * 64];

  const int lane = threadIdx.x & 31;
  const int wave = threadIdx.x >> 5;
  const int ln   = lane & 15;
  const int half = lane >> 4;
  const int kg   = half * 8;

  const int mtile  = blockIdx.x * 128 + wave * 16;
  const int n_base = blockIdx.y * 64;

  const _Float16* arow = Attn + (size_t)(mtile + ln) * DMODEL + kg;
  const int cr = threadIdx.x >> 2;
  const int cp = (threadIdx.x & 3) * 16;
  const _Float16* wsrc = W + (size_t)(n_base + cr) * DMODEL + cp;

  v8f acc[4] = {v8f{}, v8f{}, v8f{}, v8f{}};

  async_copy_b128(&BT[0][cr * 64 + cp],     wsrc);
  async_copy_b128(&BT[0][cr * 64 + cp + 8], wsrc + 8);

  for (int k = 0; k < DMODEL; k += 64) {
    const int cur = (k >> 6) & 1;
    wait_async0();
    __syncthreads();
    if (k + 64 < DMODEL) {
      async_copy_b128(&BT[cur ^ 1][cr * 64 + cp],     wsrc + k + 64);
      async_copy_b128(&BT[cur ^ 1][cr * 64 + cp + 8], wsrc + k + 64 + 8);
    }
#pragma unroll
    for (int ks = 0; ks < 64; ks += 32) {
      v16h a = frag_f16(arow + k + ks);
#pragma unroll
      for (int j = 0; j < 4; ++j)
        acc[j] = wmma_f16(a, frag_f16(&BT[cur][(16 * j + ln) * 64 + ks + kg]), acc[j]);
    }
  }

  const int mrow = mtile + half * 8;
#pragma unroll
  for (int j = 0; j < 4; ++j) {
    float bj = bias[n_base + 16 * j + ln];
#pragma unroll
    for (int r = 0; r < 8; ++r)
      out[(size_t)(mrow + r) * DMODEL + n_base + 16 * j + ln] = acc[j][r] + bj;
  }
}

// ---------------------------------------------------------------------------
// Launch
// ---------------------------------------------------------------------------
extern "C" void kernel_launch(void* const* d_in, const int* in_sizes, int n_in,
                              void* d_out, int out_size, void* d_ws, size_t ws_size,
                              hipStream_t stream) {
  (void)in_sizes; (void)n_in; (void)out_size; (void)ws_size;

  const float* X      = (const float*)d_in[0];  // hidden_states [B,S,D]
  const float* Wqkv_w = (const float*)d_in[1];  // [3D, D]
  const float* Wqkv_b = (const float*)d_in[2];  // [3D]
  const float* Wo_w   = (const float*)d_in[3];  // [D, D]
  const float* Wo_b   = (const float*)d_in[4];  // [D]
  float* out          = (float*)d_out;

  char* ws = (char*)d_ws;
  size_t off = 0;
  float* cosT = (float*)(ws + off); off += (size_t)SEQ * 32 * sizeof(float);
  float* sinT = (float*)(ws + off); off += (size_t)SEQ * 32 * sizeof(float);
  const size_t headElems = (size_t)BATCH * NHEAD * SEQ * HDIM;  // 8.39M
  _Float16* Qb = (_Float16*)(ws + off); off += headElems * sizeof(_Float16);
  _Float16* Kb = (_Float16*)(ws + off); off += headElems * sizeof(_Float16);
  _Float16* Vt = (_Float16*)(ws + off); off += headElems * sizeof(_Float16);
  _Float16* At = (_Float16*)(ws + off); off += (size_t)MTOT * DMODEL * sizeof(_Float16);
  _Float16* Xh = (_Float16*)(ws + off); off += (size_t)MTOT * DMODEL * sizeof(_Float16);
  _Float16* Wqkvh = (_Float16*)(ws + off); off += (size_t)3 * DMODEL * DMODEL * sizeof(_Float16);
  _Float16* Woh   = (_Float16*)(ws + off); off += (size_t)DMODEL * DMODEL * sizeof(_Float16);

  // f32 -> f16 operand pre-conversion (one pass, L2 resident)
  {
    int n4;
    n4 = MTOT * DMODEL / 4;
    cvt_f16_kernel<<<(n4 + 255) / 256, 256, 0, stream>>>(X, Xh, n4);
    n4 = 3 * DMODEL * DMODEL / 4;
    cvt_f16_kernel<<<(n4 + 255) / 256, 256, 0, stream>>>(Wqkv_w, Wqkvh, n4);
    n4 = DMODEL * DMODEL / 4;
    cvt_f16_kernel<<<(n4 + 255) / 256, 256, 0, stream>>>(Wo_w, Woh, n4);
  }

  rope_table_kernel<<<(SEQ * 32 + 255) / 256, 256, 0, stream>>>(cosT, sinT);

  qkv_rope_kernel<<<dim3(MTOT / 128, (3 * DMODEL) / 64), 256, 0, stream>>>(
      Xh, Wqkvh, Wqkv_b, cosT, sinT, Qb, Kb, Vt);

  attn_kernel<<<dim3(SEQ / 128, BATCH * NHEAD), 256, 0, stream>>>(Qb, Kb, Vt, At);

  out_proj_kernel<<<dim3(MTOT / 128, DMODEL / 64), 256, 0, stream>>>(At, Woh, Wo_b, out);
}
